// Attention_36850819399769
// MI455X (gfx1250) — matrix operand order repaired and checked by reading, not compiled, not run on hardware
//
#include <hip/hip_runtime.h>

// ---------------- problem constants ----------------
#define Bq   4
#define Nq   2048
#define DIMq 1024
#define Hq   16
#define HDq  64
#define SCALEq 0.125f            // 1/sqrt(64)

typedef __attribute__((ext_vector_type(16))) __bf16 v16bf;
typedef __attribute__((ext_vector_type(8)))  float  v8f;
typedef __attribute__((ext_vector_type(2)))  __bf16 bf16x2;
typedef __attribute__((ext_vector_type(2)))  float  f32x2;

__device__ __forceinline__ v8f wmma_bf16(v16bf a, v16bf b, v8f c) {
  // D = A(16x32 bf16) * B(32x16 bf16) + C(16x16 f32)
  return __builtin_amdgcn_wmma_f32_16x16x32_bf16(
      /*neg_a=*/false, a, /*neg_b=*/false, b,
      /*c_mod=*/(short)0, c, /*reuse_a=*/false, /*reuse_b=*/false);
}

// CDNA5 async DMA: global -> LDS, tracked by ASYNCcnt (CDNA5 ISA).
// VDST VGPR = LDS byte offset (low 32 bits of generic shared pointer),
// VADDR = 64-bit global address.
__device__ __forceinline__ void async_g2l_b128(unsigned int lds_off, const void* gptr) {
  asm volatile("global_load_async_to_lds_b128 %0, %1, off"
               :: "v"(lds_off), "v"(gptr) : "memory");
}
__device__ __forceinline__ unsigned int lds_off_of(const void* p) {
  return (unsigned int)(unsigned long long)p;   // LDS byte offset = low 32 bits of the pointer
}

// ============================================================
// Kernel 1: QKV projection  qkv[bn, e] = sum_d x[bn,d] * w[e,d] + bias[e]
// Each wave: 32 (bn) x 64 (e) tile, K in steps of 32, 8 WMMA / step.
// Epilogue scatters bf16 Q (scaled), K [n][hd], V^T [hd][n] per (b,h).
// ============================================================
__global__ __launch_bounds__(256) void qkv_gemm_kernel(
    const float* __restrict__ x, const float* __restrict__ w,
    const float* __restrict__ bias,
    __bf16* __restrict__ Qb, __bf16* __restrict__ Kb, __bf16* __restrict__ Vt)
{
  const int lane = threadIdx.x & 31;
  const int wid  = blockIdx.x * (blockDim.x >> 5) + (threadIdx.x >> 5);
  const int rowTile = wid & 255;         // 256 row tiles of 32 bn
  const int colTile = wid >> 8;          // 48 col tiles of 64 e
  const int row0 = rowTile * 32;
  const int col0 = colTile * 64;

  const int m16  = lane & 15;
  const int half = lane >> 4;
  const int kbA  = half * 8;             // A-frag K sub-base (ISA 16-bit A layout)
  const int kbB  = half * 8;             // B-frag K sub-base (ISA 16-bit B layout)

  v8f acc[2][4];
#pragma unroll
  for (int r = 0; r < 2; ++r)
#pragma unroll
    for (int c = 0; c < 4; ++c) acc[r][c] = (v8f){0.f,0.f,0.f,0.f,0.f,0.f,0.f,0.f};

  for (int k0 = 0; k0 < DIMq; k0 += 32) {
    v16bf a[2];
#pragma unroll
    for (int r = 0; r < 2; ++r) {
      const float* xr = x + (size_t)(row0 + r * 16 + m16) * DIMq + k0;
#pragma unroll
      for (int v = 0; v < 8; ++v) {
        const int k = (v < 4) ? (kbA + 2 * v) : (16 + kbA + 2 * (v - 4));
        f32x2 p = *reinterpret_cast<const f32x2*>(xr + k);
        a[r][2 * v]     = (__bf16)p.x;
        a[r][2 * v + 1] = (__bf16)p.y;
      }
    }
#pragma unroll
    for (int c = 0; c < 4; ++c) {
      const float* wr = w + (size_t)(col0 + c * 16 + m16) * DIMq + k0 + kbB;
      v16bf bf;
#pragma unroll
      for (int v = 0; v < 8; ++v) {
        f32x2 p = *reinterpret_cast<const f32x2*>(wr + 2 * v + (v < 4 ? 0 : 8));
        bf[2 * v]     = (__bf16)p.x;
        bf[2 * v + 1] = (__bf16)p.y;
      }
      acc[0][c] = wmma_bf16(a[0], bf, acc[0][c]);
      acc[1][c] = wmma_bf16(a[1], bf, acc[1][c]);
    }
  }

  const int t = col0 >> 10;              // 0=Q 1=K 2=V (tile-uniform)
  const int h = (col0 & 1023) >> 6;      // head (tile-uniform)
#pragma unroll
  for (int c = 0; c < 4; ++c) {
    const int e  = col0 + c * 16 + m16;
    const float bv = bias[e];
    const int hd = e & 63;
#pragma unroll
    for (int r = 0; r < 2; ++r) {
#pragma unroll
      for (int j = 0; j < 8; ++j) {
        const int row = row0 + r * 16 + j + half * 8;   // bn
        const int b   = row >> 11;
        const int n   = row & (Nq - 1);
        const float val = acc[r][c][j] + bv;
        const size_t bh = (size_t)(b * Hq + h);
        if (t == 0)      Qb[(bh * Nq + n) * HDq + hd] = (__bf16)(val * SCALEq);
        else if (t == 1) Kb[(bh * Nq + n) * HDq + hd] = (__bf16)val;
        else             Vt[(bh * HDq + hd) * Nq + n] = (__bf16)val;
      }
    }
  }
}

// ============================================================
// Kernel 2: flash attention. Block (8 waves) = one (b,h), 8 consecutive
// 16-query tiles. 32-key K/V tiles are DMA'd once per block into
// double-buffered LDS via global_load_async_to_lds_b128 (ASYNCcnt),
// overlapping the next tile's fetch with this tile's 8 WMMAs.
// ============================================================
__global__ __launch_bounds__(256) void flash_attn_kernel(
    const __bf16* __restrict__ Qb, const __bf16* __restrict__ Kb,
    const __bf16* __restrict__ Vt, float* __restrict__ out)
{
  __shared__ __align__(16) unsigned char ldsK[2][32 * HDq * 2];   // 2 x 4KB  [key][hd]
  __shared__ __align__(16) unsigned char ldsV[2][HDq * 32 * 2];   // 2 x 4KB  [hd][key]
  __shared__ unsigned short plds_raw[8 * 16 * 32];                // 8 x 1KB P tiles

  const int tid   = threadIdx.x;
  const int lane  = tid & 31;
  const int wslot = tid >> 5;
  __bf16* plds = reinterpret_cast<__bf16*>(plds_raw) + wslot * (16 * 32);

  const int bh = blockIdx.x >> 4;                    // 0..63 (b*16 + h)
  const int q0 = ((blockIdx.x & 15) * 8 + wslot) * 16;

  const int m16  = lane & 15;
  const int half = lane >> 4;
  const int kbA  = half * 8;
  const int kbB  = half * 8;

  const __bf16* Qp = Qb + (size_t)bh * Nq * HDq;
  const __bf16* Kp = Kb + (size_t)bh * Nq * HDq;
  const __bf16* Vp = Vt + (size_t)bh * HDq * Nq;

  // Staging addresses for this thread: 16B of the K tile + 16B of the V tile
  const int vhd  = tid >> 2;                         // V tile row (hd)
  const int vseg = tid & 3;                          // 16B segment within 64B row

  // Q fragments (pre-scaled), register-resident for the whole pass
  v16bf qa[2];
#pragma unroll
  for (int jc = 0; jc < 2; ++jc) {
    const __bf16* qr = Qp + (size_t)(q0 + m16) * HDq + jc * 32;
#pragma unroll
    for (int v = 0; v < 8; ++v) {
      const int k = (v < 4) ? (kbA + 2 * v) : (16 + kbA + 2 * (v - 4));
      bf16x2 p = *reinterpret_cast<const bf16x2*>(qr + k);
      qa[jc][2 * v]     = p.x;
      qa[jc][2 * v + 1] = p.y;
    }
  }

  v8f o[4];
#pragma unroll
  for (int c = 0; c < 4; ++c) o[c] = (v8f){0.f,0.f,0.f,0.f,0.f,0.f,0.f,0.f};
  float mrun[8], lrun[8];
#pragma unroll
  for (int j = 0; j < 8; ++j) { mrun[j] = -1e30f; lrun[j] = 0.f; }

  // ---- stage key-block 0 into buffer 0 ----
  {
    const char* kg = (const char*)Kp + tid * 16;                       // 4KB contiguous
    async_g2l_b128(lds_off_of(&ldsK[0][0]) + tid * 16, kg);
    const char* vg = (const char*)(Vp + (size_t)vhd * Nq) + vseg * 16; // 64B rows, 4KB stride
    async_g2l_b128(lds_off_of(&ldsV[0][0]) + tid * 16, vg);
  }

  for (int it = 0; it < Nq / 32; ++it) {
    const int cur = it & 1;
    const int kb  = it * 32;
    if (kb + 32 < Nq) {   // prefetch next tile into the other buffer
      const int nxt = cur ^ 1;
      const char* kg = (const char*)(Kp + (size_t)(kb + 32) * HDq) + tid * 16;
      async_g2l_b128(lds_off_of(&ldsK[nxt][0]) + tid * 16, kg);
      const char* vg = (const char*)(Vp + (size_t)vhd * Nq + kb + 32) + vseg * 16;
      async_g2l_b128(lds_off_of(&ldsV[nxt][0]) + tid * 16, vg);
      asm volatile("s_wait_asynccnt 0x2" ::: "memory");  // current tile's 2 DMAs done
    } else {
      asm volatile("s_wait_asynccnt 0x0" ::: "memory");
    }
    __syncthreads();   // publish current tile to all 8 waves

    const __bf16* ldsKp = reinterpret_cast<const __bf16*>(&ldsK[cur][0]);
    const __bf16* ldsVp = reinterpret_cast<const __bf16*>(&ldsV[cur][0]);

    // S tiles: keys [kb, kb+16) and [kb+16, kb+32), from LDS
    v8f s[2];
#pragma unroll
    for (int st = 0; st < 2; ++st) {
      v8f sacc = (v8f){0.f,0.f,0.f,0.f,0.f,0.f,0.f,0.f};
#pragma unroll
      for (int jc = 0; jc < 2; ++jc) {
        const __bf16* kr = ldsKp + (st * 16 + m16) * HDq + jc * 32 + kbB;
        v16bf kf;
#pragma unroll
        for (int v = 0; v < 8; ++v) {
          bf16x2 p = *reinterpret_cast<const bf16x2*>(kr + 2 * v + (v < 4 ? 0 : 8));
          kf[2 * v]     = p.x;
          kf[2 * v + 1] = p.y;
        }
        sacc = wmma_bf16(qa[jc], kf, sacc);
      }
      s[st] = sacc;
    }

    // ---- online softmax over this 32-key block ----
    float bm[8];
#pragma unroll
    for (int j = 0; j < 8; ++j) bm[j] = fmaxf(s[0][j], s[1][j]);
#pragma unroll
    for (int off = 1; off < 16; off <<= 1)
#pragma unroll
      for (int j = 0; j < 8; ++j) bm[j] = fmaxf(bm[j], __shfl_xor(bm[j], off, 32));

    float corr[8], rs[8];
#pragma unroll
    for (int j = 0; j < 8; ++j) {
      const float mn = fmaxf(mrun[j], bm[j]);
      corr[j] = __expf(mrun[j] - mn);
      const float p0 = __expf(s[0][j] - mn);
      const float p1 = __expf(s[1][j] - mn);
      mrun[j] = mn;
      rs[j] = p0 + p1;
      const int row = j + half * 8;
      plds[row * 32 + m16]      = (__bf16)p0;   // ds_store_b16
      plds[row * 32 + 16 + m16] = (__bf16)p1;
#pragma unroll
      for (int c = 0; c < 4; ++c) o[c][j] *= corr[j];
    }
#pragma unroll
    for (int off = 1; off < 16; off <<= 1)
#pragma unroll
      for (int j = 0; j < 8; ++j) rs[j] += __shfl_xor(rs[j], off, 32);
#pragma unroll
    for (int j = 0; j < 8; ++j) lrun[j] = lrun[j] * corr[j] + rs[j];

    // Same-wave LDS RAW on the P tile: explicit split-counter wait
    asm volatile("s_wait_dscnt 0x0" ::: "memory");

    // P as A-fragment (16 rows x 32 keys) from per-wave LDS
    v16bf pf;
#pragma unroll
    for (int v = 0; v < 8; ++v) {
      const int k = (v < 4) ? (kbA + 2 * v) : (16 + kbA + 2 * (v - 4));
      bf16x2 p = *reinterpret_cast<const bf16x2*>(plds + m16 * 32 + k);
      pf[2 * v]     = p.x;
      pf[2 * v + 1] = p.y;
    }

    // O += P * V   (V tile in LDS is [hd][key], contiguous key pairs)
#pragma unroll
    for (int c = 0; c < 4; ++c) {
      const __bf16* vr = ldsVp + (c * 16 + m16) * 32 + kbB;
      v16bf vf;
#pragma unroll
      for (int v = 0; v < 8; ++v) {
        bf16x2 p = *reinterpret_cast<const bf16x2*>(vr + 2 * v + (v < 4 ? 0 : 8));
        vf[2 * v]     = p.x;
        vf[2 * v + 1] = p.y;
      }
      o[c] = wmma_bf16(pf, vf, o[c]);
    }

    __syncthreads();   // all waves done reading before the buffer is re-filled
  }

  // Normalize and write out[b, q, h*HD + hd]
  const int b = bh >> 4;
  const int h = bh & 15;
#pragma unroll
  for (int c = 0; c < 4; ++c) {
#pragma unroll
    for (int j = 0; j < 8; ++j) {
      const int q  = q0 + j + half * 8;
      const int hd = c * 16 + m16;
      out[((size_t)b * Nq + q) * DIMq + h * HDq + hd] = o[c][j] / lrun[j];
    }
  }
}

// ============================================================
extern "C" void kernel_launch(void* const* d_in, const int* in_sizes, int n_in,
                              void* d_out, int out_size, void* d_ws, size_t ws_size,
                              hipStream_t stream) {
  const float* x    = (const float*)d_in[0];   // [B, N, DIM]
  const float* wqkv = (const float*)d_in[1];   // [3*DIM, DIM]
  const float* bqkv = (const float*)d_in[2];   // [3*DIM]
  float* out = (float*)d_out;                  // [B, N, DIM]

  const size_t headElems = (size_t)Bq * Hq * Nq * HDq;   // 8M elements
  __bf16* Qb = (__bf16*)d_ws;
  __bf16* Kb = Qb + headElems;
  __bf16* Vt = Kb + headElems;                 // 48 MB workspace total

  // QKV GEMM: 256 x 48 = 12288 wave tiles, 8 waves/block
  qkv_gemm_kernel<<<12288 / 8, 256, 0, stream>>>(x, wqkv, bqkv, Qb, Kb, Vt);

  // Flash attention: 64 (b,h) x 16 blocks, each block = 8 q-tiles
  flash_attn_kernel<<<1024, 256, 0, stream>>>(Qb, Kb, Vt, out);
}
